// VNDGCNN3D_3831110828781
// MI455X (gfx1250) — compile-verified, matrix operand order checked by reading, not run on hardware
//
#include <hip/hip_runtime.h>
#include <hip/hip_bf16.h>
#include <cfloat>

typedef __bf16 bf16_t;
typedef __attribute__((ext_vector_type(16))) __bf16 v16bf;
typedef __attribute__((ext_vector_type(8)))  float  v8f;

#define BATCH 4
#define NPTS  4096
#define TOTAL (BATCH * NPTS)
#define KNN   20
#define C1    64
#define C2    128
#define C3    256
#define NCLS  40
#define MIT   4   // M-tiles swept per wave in the WMMA GEMMs

// ---------------------------------------------------------------------------
// prep: fold BN into (scale, shift), convert w2/w3 to bf16, zero pooled buf.
// st layout (floats): s1[64] t1[64] s2[128] t2[128] s3[256] t3[256]
// ---------------------------------------------------------------------------
__global__ void prep_kernel(const float* __restrict__ g1, const float* __restrict__ b1,
                            const float* __restrict__ m1, const float* __restrict__ v1,
                            const float* __restrict__ g2, const float* __restrict__ b2,
                            const float* __restrict__ m2, const float* __restrict__ v2,
                            const float* __restrict__ g3, const float* __restrict__ b3,
                            const float* __restrict__ m3, const float* __restrict__ v3,
                            const float* __restrict__ w2, const float* __restrict__ w3,
                            float* __restrict__ st, bf16_t* __restrict__ w2b,
                            bf16_t* __restrict__ w3b, float* __restrict__ pooled)
{
    const int tid = threadIdx.x;  // single block, 256 threads
    for (int c = tid; c < C1; c += 256) {
        float s = g1[c] * rsqrtf(v1[c] + 1e-5f);
        st[c] = s; st[64 + c] = b1[c] - m1[c] * s;
    }
    for (int c = tid; c < C2; c += 256) {
        float s = g2[c] * rsqrtf(v2[c] + 1e-5f);
        st[128 + c] = s; st[256 + c] = b2[c] - m2[c] * s;
    }
    for (int c = tid; c < C3; c += 256) {
        float s = g3[c] * rsqrtf(v3[c] + 1e-5f);
        st[384 + c] = s; st[640 + c] = b3[c] - m3[c] * s;
    }
    for (int i = tid; i < C2 * C1; i += 256) w2b[i] = (bf16_t)w2[i];
    for (int i = tid; i < C3 * C2; i += 256) w3b[i] = (bf16_t)w3[i];
    for (int i = tid; i < BATCH * C3; i += 256) pooled[i] = 0.0f;
}

// ---------------------------------------------------------------------------
// kNN: one wave32 per query point. Distances for all 4096 candidates cached
// in LDS, then 20 selection passes with wave shuffle max-reduction.
// ---------------------------------------------------------------------------
__global__ void knn_kernel(const float* __restrict__ x, int* __restrict__ knn_idx)
{
    __shared__ float dist[2][NPTS];
    const int lane = threadIdx.x & 31;
    const int w    = threadIdx.x >> 5;
    const int row  = blockIdx.x * 2 + w;
    const int b    = row / NPTS;

    const float xn0 = x[(size_t)row * 3 + 0];
    const float xn1 = x[(size_t)row * 3 + 1];
    const float xn2 = x[(size_t)row * 3 + 2];

    for (int j = 0; j < NPTS / 32; ++j) {
        const int m = lane + j * 32;
        const float* xm = x + ((size_t)b * NPTS + m) * 3;
        const float d0 = xm[0] - xn0, d1 = xm[1] - xn1, d2 = xm[2] - xn2;
        dist[w][m] = -(d0 * d0 + d1 * d1 + d2 * d2);  // negative sq distance
    }
    // wave-private LDS slice: DS ops are in-order per wave, no barrier needed
    for (int kk = 0; kk < KNN; ++kk) {
        float bv = -FLT_MAX; int bm = 0x7fffffff;
        for (int j = 0; j < NPTS / 32; ++j) {
            const int m = lane + j * 32;
            const float v = dist[w][m];
            if (v > bv || (v == bv && m < bm)) { bv = v; bm = m; }
        }
        for (int off = 16; off > 0; off >>= 1) {
            const float ov = __shfl_down(bv, off, 32);
            const int   om = __shfl_down(bm, off, 32);
            if (ov > bv || (ov == bv && om < bm)) { bv = ov; bm = om; }
        }
        const int winner = __shfl(bm, 0, 32);
        if (lane == (winner & 31)) dist[w][winner] = -FLT_MAX;  // exclude
        if (lane == 0) knn_idx[(size_t)row * KNN + kk] = winner;
    }
}

// ---------------------------------------------------------------------------
// Edge conv: 4 points per 256-thread block; 64 threads per point, one output
// channel each. Edge features staged in LDS. max over K of relu(bn(w1.e)).
// Writes bf16 activations for the WMMA GEMM that follows.
// ---------------------------------------------------------------------------
__global__ void edge_kernel(const float* __restrict__ x, const int* __restrict__ knn_idx,
                            const float* __restrict__ w1, const float* __restrict__ st,
                            bf16_t* __restrict__ h1)
{
    __shared__ float ef[4][KNN][6];
    __shared__ float w1s[C1 * 6];
    const int tid = threadIdx.x;
    const int c   = tid & 63;
    const int p   = tid >> 6;
    const int pn  = blockIdx.x * 4 + p;
    const int b   = pn / NPTS;

    for (int i = tid; i < C1 * 6; i += 256) w1s[i] = w1[i];

    const float xc0 = x[(size_t)pn * 3 + 0];
    const float xc1 = x[(size_t)pn * 3 + 1];
    const float xc2 = x[(size_t)pn * 3 + 2];
    if (c < KNN) {
        const int nb = knn_idx[(size_t)pn * KNN + c];
        const float* xm = x + ((size_t)b * NPTS + nb) * 3;
        ef[p][c][0] = xm[0] - xc0; ef[p][c][1] = xm[1] - xc1; ef[p][c][2] = xm[2] - xc2;
        ef[p][c][3] = xc0;         ef[p][c][4] = xc1;         ef[p][c][5] = xc2;
    }
    __syncthreads();

    float wv[6];
#pragma unroll
    for (int j = 0; j < 6; ++j) wv[j] = w1s[c * 6 + j];
    const float s = st[c], t = st[64 + c];
    float mx = 0.0f;  // relu output >= 0, so 0 is the correct identity
    for (int k = 0; k < KNN; ++k) {
        float acc = 0.0f;
#pragma unroll
        for (int j = 0; j < 6; ++j) acc = fmaf(ef[p][k][j], wv[j], acc);
        mx = fmaxf(mx, fmaxf(fmaf(acc, s, t), 0.0f));
    }
    h1[(size_t)pn * C1 + c] = (bf16_t)mx;
}

// ---------------------------------------------------------------------------
// bf16 fragment helpers (layout per CDNA5 ISA):
// lane<16 holds K {0..7,16..23}, lane>=16 holds K {8..15,24..31} of its
// row (A) / column (B^T, i.e. weight row) -> two global_load_b128 per chunk.
// ---------------------------------------------------------------------------
union FragU { v16bf v; uint4 q[2]; };

__device__ __forceinline__ v16bf load_frag(const bf16_t* __restrict__ rowp,
                                           int kc, int half)
{
    FragU f;
    f.q[0] = *reinterpret_cast<const uint4*>(rowp + kc + half * 8);
    f.q[1] = *reinterpret_cast<const uint4*>(rowp + kc + 16 + half * 8);
    return f.v;
}

// ---------------------------------------------------------------------------
// Layer 2: [16384 x 64] x [128 x 64]^T -> bn+relu -> bf16 [16384 x 128]
// Each wave keeps one B (weight) tile resident and sweeps MIT M-tiles.
// ---------------------------------------------------------------------------
__global__ void gemm2_kernel(const bf16_t* __restrict__ h1, const bf16_t* __restrict__ w2b,
                             const float* __restrict__ st, bf16_t* __restrict__ h2)
{
    const int lane = threadIdx.x & 31;
    const int wave = blockIdx.x * (blockDim.x >> 5) + (threadIdx.x >> 5);
    const int ntiles = C2 / 16;                      // 8
    const int tn     = wave % ntiles;
    const int mBase0 = (wave / ntiles) * (16 * MIT);

    const int half = lane >> 4;
    const int r    = lane & 15;

    // resident B fragments: K = 64 -> 2 chunks
    const bf16_t* brow = w2b + (size_t)(tn * 16 + r) * C1;
    const v16bf fb0 = load_frag(brow, 0, half);
    const v16bf fb1 = load_frag(brow, 32, half);

    const int   col = tn * 16 + r;
    const float s = st[128 + col], t = st[256 + col];

    for (int mi = 0; mi < MIT; ++mi) {
        const int mBase = mBase0 + mi * 16;
        const bf16_t* arow = h1 + (size_t)(mBase + r) * C1;
        if (mi + 1 < MIT)
            __builtin_prefetch(arow + 16 * C1, 0, 3);   // next M-tile A rows
        v8f acc = {};
        acc = __builtin_amdgcn_wmma_f32_16x16x32_bf16(
            false, load_frag(arow, 0, half), false, fb0, (short)0, acc, false, false);
        acc = __builtin_amdgcn_wmma_f32_16x16x32_bf16(
            false, load_frag(arow, 32, half), false, fb1, (short)0, acc, false, false);

        const int rbase = mBase + half * 8;
#pragma unroll
        for (int i = 0; i < 8; ++i) {
            const float v = fmaxf(fmaf(acc[i], s, t), 0.0f);
            h2[(size_t)(rbase + i) * C2 + col] = (bf16_t)v;
        }
    }
}

// ---------------------------------------------------------------------------
// Layer 3 fused with global max-pool: [16384 x 128] x [256 x 128]^T
// -> bn+relu -> in-register max over MIT*16 rows -> one integer atomicMax
// per lane (post-ReLU values >= 0, pooled buffer zeroed by prep).
// ---------------------------------------------------------------------------
__global__ void gemm3_kernel(const bf16_t* __restrict__ h2, const bf16_t* __restrict__ w3b,
                             const float* __restrict__ st, float* __restrict__ pooled)
{
    const int lane = threadIdx.x & 31;
    const int wave = blockIdx.x * (blockDim.x >> 5) + (threadIdx.x >> 5);
    const int ntiles = C3 / 16;                      // 16
    const int tn     = wave % ntiles;
    const int mBase0 = (wave / ntiles) * (16 * MIT);

    const int half = lane >> 4;
    const int r    = lane & 15;

    // resident B fragments: K = 128 -> 4 chunks
    const bf16_t* brow = w3b + (size_t)(tn * 16 + r) * C2;
    v16bf fb[4];
#pragma unroll
    for (int kc = 0; kc < 4; ++kc) fb[kc] = load_frag(brow, kc * 32, half);

    const int   col = tn * 16 + r;
    const float s = st[384 + col], t = st[640 + col];
    float mx = 0.0f;

    for (int mi = 0; mi < MIT; ++mi) {
        const int mBase = mBase0 + mi * 16;
        const bf16_t* arow = h2 + (size_t)(mBase + r) * C2;
        if (mi + 1 < MIT)
            __builtin_prefetch(arow + 16 * C2, 0, 3);   // next M-tile A rows
        v8f acc = {};
#pragma unroll
        for (int kc = 0; kc < 4; ++kc)
            acc = __builtin_amdgcn_wmma_f32_16x16x32_bf16(
                false, load_frag(arow, kc * 32, half), false, fb[kc],
                (short)0, acc, false, false);
#pragma unroll
        for (int i = 0; i < 8; ++i)
            mx = fmaxf(mx, fmaxf(fmaf(acc[i], s, t), 0.0f));
    }
    const int b = mBase0 / NPTS;       // whole M sweep stays in one batch
    atomicMax((int*)&pooled[b * C3 + col], __float_as_int(mx));
}

// ---------------------------------------------------------------------------
// Head: (4,256) -> fc1(128)+relu -> fc2(40). Tiny; one block.
// ---------------------------------------------------------------------------
__global__ void head_kernel(const float* __restrict__ pooled,
                            const float* __restrict__ fc1_w, const float* __restrict__ fc1_b,
                            const float* __restrict__ fc2_w, const float* __restrict__ fc2_b,
                            float* __restrict__ out)
{
    __shared__ float hh[BATCH][C2];
    const int tid = threadIdx.x;  // 256
    if (tid < C2) {
        for (int b = 0; b < BATCH; ++b) {
            float acc = fc1_b[tid];
            for (int k = 0; k < C3; ++k)
                acc = fmaf(pooled[b * C3 + k], fc1_w[tid * C3 + k], acc);
            hh[b][tid] = fmaxf(acc, 0.0f);
        }
    }
    __syncthreads();
    if (tid < BATCH * NCLS) {
        const int b = tid / NCLS, o = tid % NCLS;
        float acc = fc2_b[o];
        for (int k = 0; k < C2; ++k)
            acc = fmaf(hh[b][k], fc2_w[o * C2 + k], acc);
        out[tid] = acc;
    }
}

// ---------------------------------------------------------------------------
extern "C" void kernel_launch(void* const* d_in, const int* in_sizes, int n_in,
                              void* d_out, int out_size, void* d_ws, size_t ws_size,
                              hipStream_t stream)
{
    const float* x     = (const float*)d_in[0];
    const float* w1    = (const float*)d_in[1];
    const float* g1    = (const float*)d_in[2];
    const float* b1    = (const float*)d_in[3];
    const float* m1    = (const float*)d_in[4];
    const float* v1    = (const float*)d_in[5];
    const float* w2    = (const float*)d_in[6];
    const float* g2    = (const float*)d_in[7];
    const float* b2    = (const float*)d_in[8];
    const float* m2    = (const float*)d_in[9];
    const float* v2    = (const float*)d_in[10];
    const float* w3    = (const float*)d_in[11];
    const float* g3    = (const float*)d_in[12];
    const float* b3    = (const float*)d_in[13];
    const float* m3    = (const float*)d_in[14];
    const float* v3    = (const float*)d_in[15];
    const float* fc1_w = (const float*)d_in[16];
    const float* fc1_b = (const float*)d_in[17];
    const float* fc2_w = (const float*)d_in[18];
    const float* fc2_b = (const float*)d_in[19];

    char* ws = (char*)d_ws;
    size_t off = 0;
    auto take = [&](size_t bytes) -> char* {
        char* p = ws + off;
        off += (bytes + 255) & ~(size_t)255;
        return p;
    };
    int*    idx    = (int*)   take((size_t)TOTAL * KNN * sizeof(int));
    bf16_t* h1     = (bf16_t*)take((size_t)TOTAL * C1 * sizeof(bf16_t));
    bf16_t* h2     = (bf16_t*)take((size_t)TOTAL * C2 * sizeof(bf16_t));
    bf16_t* w2b    = (bf16_t*)take((size_t)C2 * C1 * sizeof(bf16_t));
    bf16_t* w3b    = (bf16_t*)take((size_t)C3 * C2 * sizeof(bf16_t));
    float*  st     = (float*) take(896 * sizeof(float));
    float*  pooled = (float*) take((size_t)BATCH * C3 * sizeof(float));

    prep_kernel<<<1, 256, 0, stream>>>(g1, b1, m1, v1, g2, b2, m2, v2,
                                       g3, b3, m3, v3, w2, w3,
                                       st, w2b, w3b, pooled);

    knn_kernel<<<TOTAL / 2, 64, 0, stream>>>(x, idx);

    edge_kernel<<<TOTAL / 4, 256, 0, stream>>>(x, idx, w1, st, h1);

    // 4 waves (128 threads) per block; each wave sweeps MIT M-tiles
    gemm2_kernel<<<(TOTAL / (16 * MIT)) * (C2 / 16) / 4, 128, 0, stream>>>(h1, w2b, st, h2);

    gemm3_kernel<<<(TOTAL / (16 * MIT)) * (C3 / 16) / 4, 128, 0, stream>>>(h2, w3b, st, pooled);

    head_kernel<<<1, 256, 0, stream>>>(pooled, fc1_w, fc1_b, fc2_w, fc2_b, (float*)d_out);
}
